// MultiHeadAttention_76381698392401
// MI455X (gfx1250) — compile-verified, matrix operand order checked by reading, not run on hardware
//
#include <hip/hip_runtime.h>
#include <hip/hip_bf16.h>
#include <cstdint>

typedef __attribute__((ext_vector_type(16))) _Float16 v16h;
typedef __attribute__((ext_vector_type(8)))  float    v8f;

#define S_LEN  2048
#define BATCH  2
#define EMB    1024
#define NHEAD  16
#define HDIM   64
#define MROWS  (S_LEN * BATCH)   // 4096
#define NQKV   (3 * EMB)         // 3072

#define BM 128
#define BN 128
#define BK 32

// ---------------------------------------------------------------------------
// CDNA5 async DMA helpers (gfx1250): global -> LDS, tracked with ASYNCcnt.
// VFLAT form: global_load_async_to_lds_b128 vdst(lds byte addr), v[a:a+1], off
// LDS byte offset = low 32 bits of the generic pointer (LDS aperture).
// ---------------------------------------------------------------------------
__device__ __forceinline__ unsigned lds_off(const void* p) {
    return (unsigned)(uintptr_t)p;
}
__device__ __forceinline__ void async_load_b128(unsigned ldsaddr, const void* gaddr) {
    asm volatile("global_load_async_to_lds_b128 %0, %1, off"
                 :: "v"(ldsaddr), "v"((unsigned long long)(uintptr_t)gaddr)
                 : "memory");
}
template <int N>
__device__ __forceinline__ void wait_asynccnt() {
    asm volatile("s_wait_asynccnt %0" :: "i"(N) : "memory");
}

// ---------------------------------------------------------------------------
// WMMA fragment helpers (wave32, v_wmma_f32_16x16x32_f16)
// A (16x32 MxK, f16): lane L row m=L%16; pair p -> K per ISA 7.12.2.
// B (32x16 KxN): lane L column n=L%16, same K mapping.
// C/D (16x16 f32): lane L column n=L%16, rows m = v + 8*(L>=16).
// ---------------------------------------------------------------------------
__device__ __forceinline__ int pair_k(int p, int laneHi) {
    return ((p < 4) ? (2 * p) : (16 + 2 * (p - 4))) + (laneHi ? 8 : 0);
}
__device__ __forceinline__ v16h load_frag_contig(const _Float16* src, int laneHi) {
    v16h f;
#pragma unroll
    for (int p = 0; p < 8; ++p) {
        int k = pair_k(p, laneHi);
        f[2 * p]     = src[k];
        f[2 * p + 1] = src[k + 1];
    }
    return f;
}
__device__ __forceinline__ v16h load_frag_strided(const _Float16* src, int stride, int laneHi) {
    v16h f;
#pragma unroll
    for (int p = 0; p < 8; ++p) {
        int k = pair_k(p, laneHi);
        f[2 * p]     = src[k * stride];
        f[2 * p + 1] = src[(k + 1) * stride];
    }
    return f;
}
__device__ __forceinline__ v8f wmma16(v16h a, v16h b, v8f c) {
    return __builtin_amdgcn_wmma_f32_16x16x32_f16(
        false, a, false, b, (short)0, c, false, false);
}
__device__ __forceinline__ float redmax16(float x) {
#pragma unroll
    for (int m = 1; m < 16; m <<= 1) x = fmaxf(x, __shfl_xor(x, m, 32));
    return x;
}
__device__ __forceinline__ float redsum16(float x) {
#pragma unroll
    for (int m = 1; m < 16; m <<= 1) x += __shfl_xor(x, m, 32);
    return x;
}

// ---------------------------------------------------------------------------
// fp32 -> fp16 conversion
// ---------------------------------------------------------------------------
__global__ void cvt_f32_to_f16(const float* __restrict__ in,
                               _Float16* __restrict__ out, int n) {
    int i = blockIdx.x * blockDim.x + threadIdx.x;
    if (i < n) out[i] = (_Float16)in[i];
}

// ---------------------------------------------------------------------------
// Shared GEMM core: block = 8 waves (256 thr), tile 128m x 128n, BK=32.
// Wave (wm = wave>>1, wn = wave&1) computes 32m x 64n = 8 C fragments.
// A [BM][BK] and W-rows [BN][BK] double-buffered in LDS via async DMA.
// Per thread per stage: 4x global_load_async_to_lds_b128 (64 B).
// ---------------------------------------------------------------------------
struct GemmCtx {
    int tid, laneLo, laneHi, wm, wn, mbase, nbase;
};

__device__ __forceinline__ void gemm_stage(const _Float16* __restrict__ Xh,
                                           const _Float16* __restrict__ Wh,
                                           _Float16* sA, _Float16* sB,
                                           int tid, int mbase, int nbase, int kb) {
#pragma unroll
    for (int i = 0; i < 2; ++i) {
        int c   = tid + i * 256;        // 512 chunks of 16B per tile
        int row = c >> 2;               // 0..127
        int c8  = (c & 3) * 8;          // half offset within 32-half row
        async_load_b128(lds_off(sA + c * 8),
                        Xh + (size_t)(mbase + row) * EMB + kb + c8);
        async_load_b128(lds_off(sB + c * 8),
                        Wh + (size_t)(nbase + row) * EMB + kb + c8);
    }
}

__device__ __forceinline__ void gemm_compute(const _Float16* sA, const _Float16* sB,
                                             const GemmCtx& g, v8f acc[2][4]) {
    v16h af0 = load_frag_contig(sA + (g.wm * 32 + g.laneLo) * BK, g.laneHi);
    v16h af1 = load_frag_contig(sA + (g.wm * 32 + 16 + g.laneLo) * BK, g.laneHi);
    v16h bf[4];
#pragma unroll
    for (int t = 0; t < 4; ++t)
        bf[t] = load_frag_contig(sB + (g.wn * 64 + t * 16 + g.laneLo) * BK, g.laneHi);
#pragma unroll
    for (int t = 0; t < 4; ++t) acc[0][t] = wmma16(af0, bf[t], acc[0][t]);
#pragma unroll
    for (int t = 0; t < 4; ++t) acc[1][t] = wmma16(af1, bf[t], acc[1][t]);
}

#define GEMM_MAIN_LOOP(Xh, Wh)                                                 \
    __shared__ __align__(16) _Float16 sA[2][BM * BK];                          \
    __shared__ __align__(16) _Float16 sB[2][BN * BK];                          \
    GemmCtx g;                                                                 \
    g.tid = threadIdx.x;                                                       \
    int lane = g.tid & 31, wave = g.tid >> 5;                                  \
    g.laneLo = lane & 15; g.laneHi = lane >> 4;                                \
    g.wm = wave >> 1; g.wn = wave & 1;                                         \
    g.mbase = blockIdx.y * BM; g.nbase = blockIdx.x * BN;                      \
    v8f acc[2][4] = {};                                                        \
    const int nk = EMB / BK;                                                   \
    gemm_stage(Xh, Wh, sA[0], sB[0], g.tid, g.mbase, g.nbase, 0);              \
    for (int it = 0; it < nk - 1; ++it) {                                      \
        __syncthreads(); /* prior reads of next buf done */                    \
        gemm_stage(Xh, Wh, sA[(it + 1) & 1], sB[(it + 1) & 1],                 \
                   g.tid, g.mbase, g.nbase, (it + 1) * BK);                    \
        wait_asynccnt<4>(); /* current tile's 4 DMAs done; next 4 in flight */ \
        __syncthreads();                                                       \
        gemm_compute(sA[it & 1], sB[it & 1], g, acc);                          \
    }                                                                          \
    wait_asynccnt<0>();                                                        \
    __syncthreads();                                                           \
    gemm_compute(sA[(nk - 1) & 1], sB[(nk - 1) & 1], g, acc);

// ---------------------------------------------------------------------------
// QKV projection + scatter into q/k/v [B*H, S, D] f16 (q pre-scaled by 1/8).
// ---------------------------------------------------------------------------
__global__ __launch_bounds__(256) void gemm_qkv_kernel(
    const _Float16* __restrict__ Xh, const _Float16* __restrict__ Wh,
    const float* __restrict__ bias,
    _Float16* __restrict__ qh, _Float16* __restrict__ kh,
    _Float16* __restrict__ vh) {
    GEMM_MAIN_LOOP(Xh, Wh)

    const float scaling = 0.125f;  // HDIM^-0.5
#pragma unroll
    for (int ms = 0; ms < 2; ++ms) {
#pragma unroll
        for (int t = 0; t < 4; ++t) {
            int f0 = g.nbase + g.wn * 64 + t * 16;   // 16-aligned, uniform
            int f  = f0 + g.laneLo;
            int grp = f0 >> 10;                      // 0=q 1=k 2=v
            int fl  = f - (grp << 10);
            int h   = fl >> 6;
            int d   = fl & 63;
            float bv = bias[f];
            _Float16* dst = (grp == 0) ? qh : (grp == 1) ? kh : vh;
#pragma unroll
            for (int v = 0; v < 8; ++v) {
                int r = g.mbase + g.wm * 32 + ms * 16 + v + 8 * g.laneHi;
                int s = r >> 1;
                int b = r & 1;
                float val = acc[ms][t][v] + bv;
                if (grp == 0) val *= scaling;
                dst[(((size_t)(b * NHEAD + h)) * S_LEN + s) * HDIM + d] = (_Float16)val;
            }
        }
    }
}

// ---------------------------------------------------------------------------
// Output projection: fp32 + bias straight to d_out [MROWS, EMB].
// ---------------------------------------------------------------------------
__global__ __launch_bounds__(256) void gemm_out_kernel(
    const _Float16* __restrict__ Ah, const _Float16* __restrict__ Wh,
    const float* __restrict__ bias, float* __restrict__ out) {
    GEMM_MAIN_LOOP(Ah, Wh)

#pragma unroll
    for (int ms = 0; ms < 2; ++ms) {
#pragma unroll
        for (int t = 0; t < 4; ++t) {
            int f = g.nbase + g.wn * 64 + t * 16 + g.laneLo;
            float bv = bias[f];
#pragma unroll
            for (int v = 0; v < 8; ++v) {
                int r = g.mbase + g.wm * 32 + ms * 16 + v + 8 * g.laneHi;
                out[(size_t)r * EMB + f] = acc[ms][t][v] + bv;
            }
        }
    }
}

// ---------------------------------------------------------------------------
// Flash attention: block = 4 waves = 64 query rows of one (b,h).
// K/V tiles (32 keys x 64 d) DMA'd to LDS (async); online softmax in regs;
// P bounced through per-wave LDS into an A fragment for the P*V WMMAs.
// ---------------------------------------------------------------------------
__global__ __launch_bounds__(128) void flash_attn_kernel(
    const _Float16* __restrict__ qh, const _Float16* __restrict__ kh,
    const _Float16* __restrict__ vh, _Float16* __restrict__ attnh) {
    __shared__ __align__(16) _Float16 sK[32 * HDIM];
    __shared__ __align__(16) _Float16 sV[32 * HDIM];
    __shared__ __align__(16) _Float16 sP[4][16 * 32];

    const int lane   = threadIdx.x & 31;
    const int wave   = threadIdx.x >> 5;
    const int laneLo = lane & 15;
    const int laneHi = lane >> 4;
    const int bh     = blockIdx.y;      // b*NHEAD + h
    const int b      = bh >> 4;
    const int h      = bh & 15;
    const int qrow0  = blockIdx.x * 64 + wave * 16;

    const _Float16* qbase = qh + (size_t)bh * S_LEN * HDIM;
    const _Float16* kbase = kh + (size_t)bh * S_LEN * HDIM;
    const _Float16* vbase = vh + (size_t)bh * S_LEN * HDIM;

    v16h aq0 = load_frag_contig(qbase + (qrow0 + laneLo) * HDIM + 0,  laneHi);
    v16h aq1 = load_frag_contig(qbase + (qrow0 + laneLo) * HDIM + 32, laneHi);

    v8f o[4] = {};
    float mrun[8], lrun[8];
#pragma unroll
    for (int v = 0; v < 8; ++v) { mrun[v] = -__builtin_inff(); lrun[v] = 0.0f; }

    for (int j = 0; j < S_LEN; j += 32) {
        // Async-DMA the 4KB K and V tiles: 128 thr x 2 chunks x 16B per tile.
        const _Float16* gK = kbase + (size_t)j * HDIM;
        const _Float16* gV = vbase + (size_t)j * HDIM;
#pragma unroll
        for (int i = 0; i < 2; ++i) {
            int c = threadIdx.x + i * 128;          // 256 chunks per tile
            async_load_b128(lds_off(sK + c * 8), gK + c * 8);
            async_load_b128(lds_off(sV + c * 8), gV + c * 8);
        }
        wait_asynccnt<0>();
        __syncthreads();

        v8f sc0 = {}, sc1 = {};
        {
            v16h bk = load_frag_contig(sK + laneLo * HDIM + 0, laneHi);
            sc0 = wmma16(aq0, bk, sc0);
            bk = load_frag_contig(sK + laneLo * HDIM + 32, laneHi);
            sc0 = wmma16(aq1, bk, sc0);
        }
        {
            v16h bk = load_frag_contig(sK + (16 + laneLo) * HDIM + 0, laneHi);
            sc1 = wmma16(aq0, bk, sc1);
            bk = load_frag_contig(sK + (16 + laneLo) * HDIM + 32, laneHi);
            sc1 = wmma16(aq1, bk, sc1);
        }

        _Float16* sPw = sP[wave];
#pragma unroll
        for (int v = 0; v < 8; ++v) {
            float cm    = redmax16(fmaxf(sc0[v], sc1[v]));
            float mnew  = fmaxf(mrun[v], cm);
            float alpha = __expf(mrun[v] - mnew);
            float p0 = __expf(sc0[v] - mnew);
            float p1 = __expf(sc1[v] - mnew);
            float rs = redsum16(p0 + p1);
            lrun[v] = lrun[v] * alpha + rs;
            mrun[v] = mnew;
            o[0][v] *= alpha; o[1][v] *= alpha;
            o[2][v] *= alpha; o[3][v] *= alpha;
            int m = v + 8 * laneHi;
            sPw[m * 32 + laneLo]      = (_Float16)p0;
            sPw[m * 32 + 16 + laneLo] = (_Float16)p1;
        }
        __syncthreads();

        v16h ap = load_frag_contig(sPw + laneLo * 32, laneHi);
#pragma unroll
        for (int t = 0; t < 4; ++t) {
            v16h bv = load_frag_strided(sV + t * 16 + laneLo, HDIM, laneHi);
            o[t] = wmma16(ap, bv, o[t]);
        }
        __syncthreads();
    }

#pragma unroll
    for (int t = 0; t < 4; ++t) {
        int col = h * HDIM + t * 16 + laneLo;
#pragma unroll
        for (int v = 0; v < 8; ++v) {
            int srow = qrow0 + v + 8 * laneHi;
            attnh[((size_t)srow * BATCH + b) * EMB + col] =
                (_Float16)(o[t][v] / lrun[v]);
        }
    }
}

// ---------------------------------------------------------------------------
// Launch
// ---------------------------------------------------------------------------
extern "C" void kernel_launch(void* const* d_in, const int* in_sizes, int n_in,
                              void* d_out, int out_size, void* d_ws, size_t ws_size,
                              hipStream_t stream) {
    const float* x     = (const float*)d_in[0];
    const float* w_in  = (const float*)d_in[1];
    const float* b_in  = (const float*)d_in[2];
    const float* w_out = (const float*)d_in[3];
    const float* b_out = (const float*)d_in[4];
    float* out = (float*)d_out;

    // workspace layout (fp16 elements); total = 24M halves = 48 MB
    _Float16* ws = (_Float16*)d_ws;
    size_t off = 0;
    _Float16* xh  = ws + off; off += (size_t)MROWS * EMB;
    _Float16* wih = ws + off; off += (size_t)NQKV * EMB;
    _Float16* woh = ws + off; off += (size_t)EMB * EMB;
    _Float16* qh  = ws + off; off += (size_t)BATCH * NHEAD * S_LEN * HDIM;
    _Float16* kh  = ws + off; off += (size_t)BATCH * NHEAD * S_LEN * HDIM;
    _Float16* vh  = ws + off; off += (size_t)BATCH * NHEAD * S_LEN * HDIM;
    _Float16* ah  = ws + off; off += (size_t)MROWS * EMB;

    int nx = MROWS * EMB;
    cvt_f32_to_f16<<<(nx + 255) / 256, 256, 0, stream>>>(x, xh, nx);
    int nwi = NQKV * EMB;
    cvt_f32_to_f16<<<(nwi + 255) / 256, 256, 0, stream>>>(w_in, wih, nwi);
    int nwo = EMB * EMB;
    cvt_f32_to_f16<<<(nwo + 255) / 256, 256, 0, stream>>>(w_out, woh, nwo);

    gemm_qkv_kernel<<<dim3(NQKV / BN, MROWS / BM), 256, 0, stream>>>(
        xh, wih, b_in, qh, kh, vh);

    flash_attn_kernel<<<dim3(S_LEN / 64, BATCH * NHEAD), 128, 0, stream>>>(
        qh, kh, vh, ah);

    gemm_out_kernel<<<dim3(EMB / BN, MROWS / BM), 256, 0, stream>>>(
        ah, woh, b_out, out);
}
